// Attention_2104533975440
// MI455X (gfx1250) — matrix operand order repaired and checked by reading, not compiled, not run on hardware
//
#include <hip/hip_runtime.h>
#include <hip/hip_bf16.h>

// ---------------------------------------------------------------------------
// Attention forward for MI455X (gfx1250, wave32, WMMA + TDM).
//   x:[2,2048,1024] f32, mask:[1,1,2048,2048], bias:[2,16,2048,2048],
//   wq/wk/wv/wo:[1024,1024] f32.  out:[2,2048,1024] f32.
// Pipeline: GEMM(x,wq)->Q bf16 ; GEMM(x,wk)->K^T bf16 [b,h,d,s] ;
//           GEMM(x,wv)->V bf16 ; flash attention (bf16 WMMA, f32 softmax) ;
//           GEMM(attn,wo)->out f32 (A tile staged to LDS via TDM).
// Roofline: ~69 GFLOP bf16-WMMA math vs 537MB attn_bias stream (~23us at
// 23.3 TB/s) -> attention kernel is bias-bandwidth-bound; bias uses NT loads.
// ---------------------------------------------------------------------------

#define DIM 1024
#define NH  16
#define HD  64
#define BB  2
#define SS  2048

typedef __bf16 bf16;
typedef __attribute__((ext_vector_type(16))) __bf16 v16bf;
typedef __attribute__((ext_vector_type(8)))  __bf16 v8bf;
typedef __attribute__((ext_vector_type(8)))  float  v8f;

#if defined(__has_builtin)
#if __has_builtin(__builtin_amdgcn_tensor_load_to_lds) && \
    __has_builtin(__builtin_amdgcn_s_wait_tensorcnt)
#define HAVE_TDM 1
#endif
#endif
#ifndef HAVE_TDM
#define HAVE_TDM 0
#endif

#if HAVE_TDM
typedef unsigned int u32x4 __attribute__((ext_vector_type(4)));
typedef int          i32x4 __attribute__((ext_vector_type(4)));
typedef int          i32x8 __attribute__((ext_vector_type(8)));
#endif

__device__ __forceinline__ v8bf cvt8(float4 a, float4 b) {
  v8bf r;
  r[0]=(bf16)a.x; r[1]=(bf16)a.y; r[2]=(bf16)a.z; r[3]=(bf16)a.w;
  r[4]=(bf16)b.x; r[5]=(bf16)b.y; r[6]=(bf16)b.z; r[7]=(bf16)b.w;
  return r;
}

__device__ __forceinline__ v16bf cat16(v8bf lo, v8bf hi) {
  v16bf r;
#pragma unroll
  for (int i = 0; i < 8; ++i) { r[i] = lo[i]; r[i + 8] = hi[i]; }
  return r;
}

// ---------------------------------------------------------------------------
// GEMM: C[m][n] = sum_k A[m][k] * W[n][k]   (M=4096, N=K=1024)
// Block tile 64(M) x 128(N), 256 threads = 8 waves (4 along M x 2 along N),
// each wave: 16x64 output = 4 accumulators, K-step 32 via LDS bf16 tiles.
// ABF16: A tile is bf16 in memory -> staged with TDM tensor_load_to_lds.
// MODE 0: store bf16 [m][n]        (Q, V)
// MODE 1: store bf16 K^T [b,h,d,s] (K)
// MODE 2: store f32  [m][n]        (final projection)
// ---------------------------------------------------------------------------
template <bool ABF16, int MODE>
__global__ void __launch_bounds__(256)
gemm_wmma_k(const void* __restrict__ Av, const float* __restrict__ W,
            void* __restrict__ Out) {
  __shared__ bf16 As[64 * 32];    // [m][k]
  __shared__ bf16 Bs[32 * 128];   // [k][n]  (transposed from W)

  const int tid  = threadIdx.x;
  const int m0   = blockIdx.y * 64;
  const int n0   = blockIdx.x * 128;
  const int lane = tid & 31;
  const int wave = tid >> 5;
  const int wm   = (wave & 3) * 16;   // wave M offset in block tile
  const int wn   = (wave >> 2) * 64;  // wave N offset in block tile
  const int r    = lane & 15;
  const int hlf  = lane >> 4;

  v8f acc[4] = {};

  // staging assignments
  const int arow = tid >> 2;         // 0..63
  const int akc  = (tid & 3) * 8;    // k chunk 0/8/16/24
  const int bn   = (tid & 63) * 2;   // n pair 0..126
  const int bk   = (tid >> 6) * 8;   // k chunk 0/8/16/24

  for (int k0 = 0; k0 < DIM; k0 += 32) {
    // ---- stage A tile (64x32 bf16) ----
    bool manualA = true;
#if HAVE_TDM
    if (ABF16) {
      manualA = false;
      if (wave == 0) {
        // Tensor DMA: 2D tile 32(x) x 64(rows) of bf16 from [4096,1024]
        // tensor into As (row-major, contiguous) -- one DMA per workgroup.
        const bf16* gA = (const bf16*)Av + (size_t)m0 * DIM + k0;
        unsigned long long ga = (unsigned long long)(const void*)gA;
        unsigned ldsa = (unsigned)(size_t)(void*)As;
        u32x4 g0;
        g0[0] = 1u;                                    // count=1 (valid D#)
        g0[1] = ldsa;                                  // lds_addr
        g0[2] = (unsigned)(ga & 0xffffffffull);        // global_addr[31:0]
        g0[3] = (unsigned)((ga >> 32) & 0x1ffffffull)  // global_addr[56:32]
                | (2u << 30);                          // type=2 ("image")
        i32x8 g1;
        g1[0] = 1 << 16;                               // data_size=1 -> 2B
        g1[1] = (DIM & 0xffff) << 16;                  // tensor_dim0 lo16
        g1[2] = ((DIM >> 16) & 0xffff)                 // tensor_dim0 hi16
                | (((BB * SS) & 0xffff) << 16);        // tensor_dim1 lo16
        g1[3] = 32 << 16;                              // tile_dim0=32
        g1[4] = 64;                                    // tile_dim1=64
        g1[5] = DIM;                                   // tensor_dim0_stride
        g1[6] = 0;
        g1[7] = 0;
        i32x4 z4 = {};
#if defined(__clang_major__) && (__clang_major__ >= 23)
        i32x8 z8 = {};
        __builtin_amdgcn_tensor_load_to_lds(g0, g1, z4, z4, z8, 0);
#else
        __builtin_amdgcn_tensor_load_to_lds(g0, g1, z4, z4, 0);
#endif
      }
    }
#endif
    if (manualA) {
      if (ABF16) {
        const bf16* A = (const bf16*)Av;
        v8bf v = *(const v8bf*)(A + (size_t)(m0 + arow) * DIM + k0 + akc);
        *(v8bf*)(As + arow * 32 + akc) = v;
      } else {
        const float* A = (const float*)Av;
        const float* p = A + (size_t)(m0 + arow) * DIM + k0 + akc;
        float4 f0 = *(const float4*)(p);
        float4 f1 = *(const float4*)(p + 4);
        *(v8bf*)(As + arow * 32 + akc) = cvt8(f0, f1);
      }
    }
    // ---- stage B tile (32x128 bf16) transposed from W[n][k] ----
    {
      const float* w0 = W + (size_t)(n0 + bn) * DIM + k0 + bk;
      const float* w1 = w0 + DIM;
      float4 a0 = *(const float4*)(w0);
      float4 a1 = *(const float4*)(w0 + 4);
      float4 c0 = *(const float4*)(w1);
      float4 c1 = *(const float4*)(w1 + 4);
      v8bf lo = cvt8(a0, a1);   // n = bn,   k = bk..bk+7
      v8bf hi = cvt8(c0, c1);   // n = bn+1, k = bk..bk+7
#pragma unroll
      for (int i = 0; i < 8; ++i) {
        unsigned short ul = __builtin_bit_cast(unsigned short, (bf16)lo[i]);
        unsigned short uh = __builtin_bit_cast(unsigned short, (bf16)hi[i]);
        *(unsigned int*)(Bs + (bk + i) * 128 + bn) =
            (unsigned int)ul | ((unsigned int)uh << 16);
      }
    }
#if HAVE_TDM
    if (ABF16) __builtin_amdgcn_s_wait_tensorcnt(0);   // TDM LDS writes done
#endif
    __syncthreads();

    // ---- compute: 1 A frag, 4 B frags, 4 WMMA ----
    v16bf af;
    {
      const bf16* ab = As + (wm + r) * 32 + hlf * 8;
      af = cat16(*(const v8bf*)(ab), *(const v8bf*)(ab + 16));
    }
#pragma unroll
    for (int f = 0; f < 4; ++f) {
      v16bf bfv;
#pragma unroll
      for (int i = 0; i < 16; ++i)
        bfv[i] = Bs[((i & 7) + hlf * 8 + (i >> 3) * 16) * 128 + wn + f * 16 + r];
      acc[f] = __builtin_amdgcn_wmma_f32_16x16x32_bf16(
          false, af, false, bfv, (short)0, acc[f], false, false);
    }
    __syncthreads();
  }

  // ---- epilogue: C layout: VGPR j -> row wm+j+8*hlf, col = r ----
#pragma unroll
  for (int f = 0; f < 4; ++f) {
#pragma unroll
    for (int j = 0; j < 8; ++j) {
      int m = m0 + wm + j + hlf * 8;
      int n = n0 + wn + f * 16 + r;
      float v = acc[f][j];
      if (MODE == 0) {
        ((bf16*)Out)[(size_t)m * DIM + n] = (bf16)v;
      } else if (MODE == 1) {
        int b = m >> 11, s = m & (SS - 1);
        int h = n >> 6,  d = n & (HD - 1);
        ((bf16*)Out)[((size_t)(b * NH + h) * HD + d) * SS + s] = (bf16)v;
      } else {
        ((float*)Out)[(size_t)m * DIM + n] = v;
      }
    }
  }
}

// ---------------------------------------------------------------------------
// Flash attention: one wave handles (b, h, 16 q-rows), streams kv in steps of
// 32. Scores via 4 WMMA (K=64 split in two 32-chunks x two 16-col tiles),
// bias+mask add in C layout, online softmax (16-lane shuffles), P restaged
// through LDS into A layout, P.V via 4 WMMA into 16x64 f32 accumulators.
// ---------------------------------------------------------------------------
__global__ void __launch_bounds__(256)
attn_wmma_k(const bf16* __restrict__ Q, const bf16* __restrict__ KT,
            const bf16* __restrict__ V, const float* __restrict__ bias,
            const float* __restrict__ mask, bf16* __restrict__ Oattn) {
  __shared__ bf16 Ps[8][16 * 32];   // per-wave P tile (16 q x 32 kv)

  const int tid  = threadIdx.x;
  const int lane = tid & 31;
  const int wave = tid >> 5;
  const int job  = blockIdx.x * 8 + wave;      // 4096 jobs
  const int qt   = job & 127;
  const int h    = (job >> 7) & (NH - 1);
  const int b    = job >> 11;
  const int q0   = qt * 16;
  const int r    = lane & 15;
  const int hlf  = lane >> 4;

  // --- Q fragments (A layout, K = head_dim 64 -> 2 frags) ---
  const bf16* qbase = Q + ((size_t)(b * SS + q0 + r) * DIM + h * HD);
  v16bf qf[2];
#pragma unroll
  for (int kk = 0; kk < 2; ++kk) {
    const bf16* p = qbase + kk * 32 + hlf * 8;
    qf[kk] = cat16(*(const v8bf*)(p), *(const v8bf*)(p + 16));
  }

  const bf16*  ktb   = KT + (size_t)(b * NH + h) * HD * SS;
  const bf16*  vb    = V + ((size_t)b * SS * DIM + h * HD);
  const float* biasb = bias + ((size_t)(b * NH + h) * SS + q0) * SS;
  const float* maskb = mask + (size_t)q0 * SS;

  float mrow[8], lrow[8];
  v8f   o[4] = {};
#pragma unroll
  for (int j = 0; j < 8; ++j) { mrow[j] = -3.0e38f; lrow[j] = 0.f; }

  for (int s0 = 0; s0 < SS; s0 += 32) {
    // ---- scores: 2 tiles (kv 16 each), K = 64 in two 32-chunks ----
    v8f sc[2] = {};
#pragma unroll
    for (int t = 0; t < 2; ++t) {
#pragma unroll
      for (int c = 0; c < 2; ++c) {
        v16bf kf;
#pragma unroll
        for (int i = 0; i < 16; ++i)
          kf[i] = ktb[(size_t)(c * 32 + (i & 7) + hlf * 8 + (i >> 3) * 16) * SS +
                      s0 + t * 16 + r];
        sc[t] = __builtin_amdgcn_wmma_f32_16x16x32_bf16(
            false, qf[c], false, kf, (short)0, sc[t], false, false);
      }
    }

    // ---- scale + bias + mask (C layout: row q0+j+8*hlf, col s0+t*16+r) ----
#pragma unroll
    for (int t = 0; t < 2; ++t) {
#pragma unroll
      for (int j = 0; j < 8; ++j) {
        size_t bi = (size_t)(j + hlf * 8) * SS + s0 + t * 16 + r;
        float bv = __builtin_nontemporal_load(biasb + bi);  // 537MB stream
        float mv = maskb[bi];                                // L2-resident
        sc[t][j] = sc[t][j] * 0.125f + bv + mv;
      }
    }

    // ---- online softmax ----
#pragma unroll
    for (int j = 0; j < 8; ++j) {
      float mx = fmaxf(sc[0][j], sc[1][j]);
#pragma unroll
      for (int off = 8; off > 0; off >>= 1)
        mx = fmaxf(mx, __shfl_xor(mx, off, 16));
      float mn   = fmaxf(mrow[j], mx);
      float corr = __expf(mrow[j] - mn);
      mrow[j]    = mn;
      float p0 = __expf(sc[0][j] - mn);
      float p1 = __expf(sc[1][j] - mn);
      sc[0][j] = p0; sc[1][j] = p1;
      float rs = p0 + p1;
#pragma unroll
      for (int off = 8; off > 0; off >>= 1)
        rs += __shfl_xor(rs, off, 16);
      lrow[j] = lrow[j] * corr + rs;
#pragma unroll
      for (int f = 0; f < 4; ++f) o[f][j] *= corr;
    }

    // ---- P: C layout -> LDS -> A layout (wave-private, DS in-order) ----
#pragma unroll
    for (int t = 0; t < 2; ++t)
#pragma unroll
      for (int j = 0; j < 8; ++j)
        Ps[wave][(j + hlf * 8) * 32 + t * 16 + r] = (bf16)sc[t][j];

    v16bf pa;
    {
      const bf16* pb = &Ps[wave][r * 32 + hlf * 8];
      pa = cat16(*(const v8bf*)(pb), *(const v8bf*)(pb + 16));
    }

    // ---- P(16x32) . V(32x64): 4 WMMA ----
#pragma unroll
    for (int f = 0; f < 4; ++f) {
      v16bf vf;
#pragma unroll
      for (int i = 0; i < 16; ++i)
        vf[i] = vb[(size_t)(s0 + (i & 7) + hlf * 8 + (i >> 3) * 16) * DIM + f * 16 + r];
      o[f] = __builtin_amdgcn_wmma_f32_16x16x32_bf16(
          false, pa, false, vf, (short)0, o[f], false, false);
    }
  }

  // ---- normalize and store attention output bf16 [b,s,h,d] ----
  float inv[8];
#pragma unroll
  for (int j = 0; j < 8; ++j) inv[j] = 1.f / lrow[j];
#pragma unroll
  for (int f = 0; f < 4; ++f)
#pragma unroll
    for (int j = 0; j < 8; ++j) {
      int q = q0 + j + hlf * 8;
      Oattn[(size_t)(b * SS + q) * DIM + h * HD + f * 16 + r] =
          (bf16)(o[f][j] * inv[j]);
    }
}

// ---------------------------------------------------------------------------
extern "C" void kernel_launch(void* const* d_in, const int* in_sizes, int n_in,
                              void* d_out, int out_size, void* d_ws,
                              size_t ws_size, hipStream_t stream) {
  const float* x    = (const float*)d_in[0];
  const float* mask = (const float*)d_in[1];
  const float* bias = (const float*)d_in[2];
  const float* wq   = (const float*)d_in[3];
  const float* wk   = (const float*)d_in[4];
  const float* wv   = (const float*)d_in[5];
  const float* wo   = (const float*)d_in[6];
  float* out = (float*)d_out;

  const size_t NTOK = (size_t)BB * SS * DIM;   // 4M elements
  bf16* Q  = (bf16*)d_ws;
  bf16* KT = Q + NTOK;
  bf16* V  = KT + NTOK;
  bf16* AO = V + NTOK;                          // total 32 MB of workspace

  dim3 blk(256);
  dim3 gg(DIM / 128, (BB * SS) / 64);           // (8, 64)

  gemm_wmma_k<false, 0><<<gg, blk, 0, stream>>>(x, wq, Q);
  gemm_wmma_k<false, 1><<<gg, blk, 0, stream>>>(x, wk, KT);
  gemm_wmma_k<false, 0><<<gg, blk, 0, stream>>>(x, wv, V);

  attn_wmma_k<<<dim3((BB * NH * (SS / 16)) / 8), blk, 0, stream>>>(
      Q, KT, V, bias, mask, AO);

  gemm_wmma_k<true, 2><<<gg, blk, 0, stream>>>(AO, wo, out);
}
